// Generator_40613210751552
// MI455X (gfx1250) — matrix operand order repaired and checked by reading, not compiled, not run on hardware
//
#include <hip/hip_runtime.h>
#include <hip/hip_bf16.h>
#include <math.h>

typedef __attribute__((ext_vector_type(16))) _Float16 v16h;
typedef __attribute__((ext_vector_type(8)))  _Float16 v8h;
typedef __attribute__((ext_vector_type(8)))  float    v8f;

#define BETA   0.95f
#define THRESH 1.0f

// ---------------------------------------------------------------------------
// Pack weight W [N, Kreal] (row-major f32) into WMMA-B-fragment-ordered f16.
// Packed layout: tile index (nt * numKt + kt) -> 512 f16 = 32 lanes x 16 elems.
// Element e of lane L in tile (nt,kt) = W[nt*16 + (L&15)][kt*32 + ((L>>4)?8:0) + (e&7) + ((e>>3)?16:0)]
// Zero-pads n >= N and k >= Kreal (layer-1 K pad, layer-5 N pad).
// ---------------------------------------------------------------------------
__global__ void snn_pack_w(const float* __restrict__ W, _Float16* __restrict__ Bp,
                           int N, int Kreal, int numKt, int numNt) {
    long tid = (long)blockIdx.x * blockDim.x + threadIdx.x;
    long total = (long)numNt * numKt * 512;
    if (tid >= total) return;
    int e    = (int)(tid & 15);
    int lane = (int)((tid >> 4) & 31);
    long tile = tid >> 9;
    int kt = (int)(tile % numKt);
    int nt = (int)(tile / numKt);
    int n = nt * 16 + (lane & 15);
    int k = kt * 32 + ((lane >> 4) ? 8 : 0) + (e & 7) + ((e >> 3) ? 16 : 0);
    float v = (n < N && k < Kreal) ? W[(size_t)n * Kreal + k] : 0.0f;
    Bp[tid] = (_Float16)v;
}

// Convert x [M, Kreal] f32 -> f16 [M, Kpad] row-major, zero-padded K.
__global__ void snn_pack_x(const float* __restrict__ x, _Float16* __restrict__ xp,
                           int Kreal, int Kpad, int total) {
    int i = blockIdx.x * blockDim.x + threadIdx.x;
    if (i >= total) return;
    int m = i / Kpad;
    int k = i - m * Kpad;
    xp[i] = (_Float16)(k < Kreal ? x[(size_t)m * Kreal + k] : 0.0f);
}

// Elementwise LIF for layer 1 (cur1 precomputed once, reused all 25 steps).
__global__ void snn_lif1(const float* __restrict__ cur, float* __restrict__ mem,
                         _Float16* __restrict__ spk, int n) {
    int i = blockIdx.x * blockDim.x + threadIdx.x;
    if (i >= n) return;
    float mo    = mem[i];
    float reset = (mo - THRESH > 0.0f) ? 1.0f : 0.0f;
    float mn    = BETA * mo + cur[i] - reset * THRESH;
    mem[i] = mn;
    spk[i] = (_Float16)((mn - THRESH > 0.0f) ? 1.0f : 0.0f);
}

// ---------------------------------------------------------------------------
// Fused GEMM (+ optional LIF epilogue) using v_wmma_f32_16x16x32_f16.
//   A  : [M, K] row-major f16 (spikes, exactly 0/1, or padded x)
//   Bp : packed WMMA fragments (see snn_pack_w)
// Wave tile 32x32 (2x2 WMMA tiles), block = 8 waves as 4(M) x 2(N) -> 128x64.
// mode 0: curOut[m,n]  = relu(acc + bias)                       (layer-1 pre-pass)
// mode 1: LIF update; spkOut f16 [M,N]                          (layers 2..4)
// mode 2: LIF update; outSpk f32 = spk, outMem f32 = tanh(mem)  (layer 5)
// ---------------------------------------------------------------------------
__global__ __launch_bounds__(256) void snn_gemm_lif(
    const _Float16* __restrict__ A, const _Float16* __restrict__ Bp,
    const float* __restrict__ bias, float* __restrict__ mem,
    _Float16* __restrict__ spkOut, float* __restrict__ curOut,
    float* __restrict__ outSpk, float* __restrict__ outMem,
    int M, int N, int K, int mode)
{
    const int lane  = threadIdx.x & 31;
    const int wave  = threadIdx.x >> 5;
    const int wm    = wave >> 1;          // 0..3
    const int wn    = wave & 1;           // 0..1
    const int mBase = blockIdx.y * 128 + wm * 32;
    const int nBase = blockIdx.x * 64  + wn * 32;
    const int numKt = K >> 5;
    const int lrow  = lane & 15;
    const int lhi   = lane >> 4;

    v8f c00 = {}; v8f c01 = {}; v8f c10 = {}; v8f c11 = {};

    const _Float16* arow0 = A + (size_t)(mBase + lrow) * K;
    const _Float16* arow1 = A + (size_t)(mBase + 16 + lrow) * K;
    const int kOff = lhi * 8;

    const int nt0 = nBase >> 4;
    const _Float16* bptr0 = Bp + ((size_t)nt0       * numKt) * 512 + lane * 16;
    const _Float16* bptr1 = Bp + ((size_t)(nt0 + 1) * numKt) * 512 + lane * 16;

    #pragma unroll 2
    for (int kt = 0; kt < numKt; ++kt) {
        const int k0 = kt * 32 + kOff;
        union { v16h v; v8h h[2]; } a0, a1;
        a0.h[0] = *(const v8h*)(arow0 + k0);
        a0.h[1] = *(const v8h*)(arow0 + k0 + 16);
        a1.h[0] = *(const v8h*)(arow1 + k0);
        a1.h[1] = *(const v8h*)(arow1 + k0 + 16);
        const v16h b0 = *(const v16h*)(bptr0 + (size_t)kt * 512);
        const v16h b1 = *(const v16h*)(bptr1 + (size_t)kt * 512);

        c00 = __builtin_amdgcn_wmma_f32_16x16x32_f16(false, a0.v, false, b0, (short)0, c00, false, false);
        c01 = __builtin_amdgcn_wmma_f32_16x16x32_f16(false, a0.v, false, b1, (short)0, c01, false, false);
        c10 = __builtin_amdgcn_wmma_f32_16x16x32_f16(false, a1.v, false, b0, (short)0, c10, false, false);
        c11 = __builtin_amdgcn_wmma_f32_16x16x32_f16(false, a1.v, false, b1, (short)0, c11, false, false);
    }

    // Epilogue: C/D layout -> VGPR i: lanes 0-15 (M=i, N=lane), lanes 16-31 (M=8+i, N=lane-16)
    v8f acc[2][2] = { { c00, c01 }, { c10, c11 } };
    #pragma unroll
    for (int mi = 0; mi < 2; ++mi) {
        #pragma unroll
        for (int ni = 0; ni < 2; ++ni) {
            #pragma unroll
            for (int i = 0; i < 8; ++i) {
                const int m = mBase + mi * 16 + lhi * 8 + i;
                const int n = nBase + ni * 16 + lrow;
                if (n >= N) continue;
                const float raw = acc[mi][ni][i] + bias[n];
                const float cur = raw > 0.0f ? raw : 0.0f;
                const size_t idx = (size_t)m * N + n;
                if (mode == 0) {
                    curOut[idx] = cur;
                } else {
                    const float mo    = mem[idx];
                    const float reset = (mo - THRESH > 0.0f) ? 1.0f : 0.0f;
                    const float mn    = BETA * mo + cur - reset * THRESH;
                    mem[idx] = mn;
                    const float spk = (mn - THRESH > 0.0f) ? 1.0f : 0.0f;
                    if (mode == 1) {
                        spkOut[idx] = (_Float16)spk;
                    } else {
                        outSpk[idx] = spk;
                        outMem[idx] = tanhf(mn);
                    }
                }
            }
        }
    }
}

// ---------------------------------------------------------------------------
extern "C" void kernel_launch(void* const* d_in, const int* in_sizes, int n_in,
                              void* d_out, int out_size, void* d_ws, size_t ws_size,
                              hipStream_t stream) {
    (void)in_sizes; (void)n_in; (void)out_size; (void)ws_size;

    const int B = 2048;
    const int NUM_STEPS = 25;
    // layer l: output dim d[l], input dim (padded to mult of 32) Kin[l]
    const int d[5]     = { 128, 256, 512, 1024, 784 };
    const int Kin[5]   = { 128, 128, 256, 512, 1024 };  // layer-1 K padded 100->128
    const int Kreal[5] = { 100, 128, 256, 512, 1024 };
    const int gx[5]    = { 2, 4, 8, 16, 13 };            // ceil(d/64)
    // numNt = gx*4 (16-wide n-tiles, zero-padded), numKt = Kin/32

    const float* x = (const float*)d_in[0];
    const float* W[5]; const float* bvec[5];
    for (int l = 0; l < 5; ++l) { W[l] = (const float*)d_in[1 + 2 * l]; bvec[l] = (const float*)d_in[2 + 2 * l]; }
    float* out = (float*)d_out;

    // ---- workspace carving (256B aligned regions) ----
    char* ws = (char*)d_ws;
    size_t off = 0;
    auto carve = [&](size_t bytes) { char* p = ws + off; off = (off + bytes + 255) & ~(size_t)255; return p; };

    float*    memBuf[5];
    for (int l = 0; l < 5; ++l) memBuf[l] = (float*)carve((size_t)B * d[l] * 4);
    size_t memBytes = off;                               // zero this region each call
    float*    cur1 = (float*)carve((size_t)B * 128 * 4);
    _Float16* spk[4];
    for (int l = 0; l < 4; ++l) spk[l] = (_Float16*)carve((size_t)B * d[l] * 2);
    _Float16* xpad = (_Float16*)carve((size_t)B * 128 * 2);
    _Float16* Wp[5];
    for (int l = 0; l < 5; ++l) {
        size_t elems = (size_t)(gx[l] * 4) * (Kin[l] / 32) * 512;
        Wp[l] = (_Float16*)carve(elems * 2);
    }

    // ---- deterministic init: zero membrane state ----
    hipMemsetAsync(d_ws, 0, memBytes, stream);

    // ---- pack weights + input ----
    for (int l = 0; l < 5; ++l) {
        long total = (long)(gx[l] * 4) * (Kin[l] / 32) * 512;
        snn_pack_w<<<dim3((unsigned)((total + 255) / 256)), dim3(256), 0, stream>>>(
            W[l], Wp[l], d[l], Kreal[l], Kin[l] / 32, gx[l] * 4);
    }
    {
        int total = B * 128;
        snn_pack_x<<<dim3((total + 255) / 256), dim3(256), 0, stream>>>(x, xpad, 100, 128, total);
    }

    // ---- layer-1 current: constant across timesteps, compute once ----
    snn_gemm_lif<<<dim3(gx[0], B / 128), dim3(256), 0, stream>>>(
        xpad, Wp[0], bvec[0], nullptr, nullptr, cur1, nullptr, nullptr,
        B, d[0], Kin[0], /*mode=*/0);

    const size_t stepElems = (size_t)B * 784;
    const size_t memRecOff = (size_t)NUM_STEPS * stepElems;

    for (int t = 0; t < NUM_STEPS; ++t) {
        // layer 1 LIF (elementwise)
        {
            int n = B * d[0];
            snn_lif1<<<dim3((n + 255) / 256), dim3(256), 0, stream>>>(cur1, memBuf[0], spk[0], n);
        }
        // layers 2..4: fused GEMM + LIF, spikes out as f16 A-matrix for next layer
        for (int l = 1; l < 4; ++l) {
            snn_gemm_lif<<<dim3(gx[l], B / 128), dim3(256), 0, stream>>>(
                spk[l - 1], Wp[l], bvec[l], memBuf[l], spk[l], nullptr, nullptr, nullptr,
                B, d[l], Kin[l], /*mode=*/1);
        }
        // layer 5: fused GEMM + LIF, write spk_rec and tanh(mem_rec) to d_out
        snn_gemm_lif<<<dim3(gx[4], B / 128), dim3(256), 0, stream>>>(
            spk[3], Wp[4], bvec[4], memBuf[4], nullptr, nullptr,
            out + (size_t)t * stepElems, out + memRecOff + (size_t)t * stepElems,
            B, d[4], Kin[4], /*mode=*/2);
    }
}
